// PIGAModule_67345087201286
// MI455X (gfx1250) — matrix operand order repaired and checked by reading, not compiled, not run on hardware
//
#include <hip/hip_runtime.h>
#include <hip/hip_bf16.h>
#include <stdint.h>

typedef _Float16 vh8  __attribute__((ext_vector_type(8)));
typedef _Float16 vh16 __attribute__((ext_vector_type(16)));
typedef float    vf8  __attribute__((ext_vector_type(8)));
typedef float    vf4  __attribute__((ext_vector_type(4)));

#define WMMA_F16(a, b, c) \
  __builtin_amdgcn_wmma_f32_16x16x32_f16(false, (a), false, (b), (short)0, (c), false, false)

__device__ __forceinline__ vh16 mkfrag(vh8 lo, vh8 hi) {
  return __builtin_shufflevector(lo, hi, 0,1,2,3,4,5,6,7,8,9,10,11,12,13,14,15);
}
__device__ __forceinline__ vf8 vzero8() {
  vf8 z;
#pragma unroll
  for (int i = 0; i < 8; ++i) z[i] = 0.0f;
  return z;
}

// CDNA5 async global->LDS copy (ASYNCcnt-tracked, no data through VGPRs).
// lds_off = wave-relative LDS byte offset (low 32 bits of generic shared ptr).
__device__ __forceinline__ void async_load_b128(uint32_t lds_off, const void* gptr) {
  asm volatile("global_load_async_to_lds_b128 %0, %1, off"
               :: "v"(lds_off), "v"(gptr) : "memory");
}
__device__ __forceinline__ void wait_asynccnt0() {
  asm volatile("s_wait_asynccnt 0" ::: "memory");
}

// ---------------------------------------------------------------------------
// f32 -> f16 conversion (x and all weights)
// ---------------------------------------------------------------------------
__global__ void f32_to_f16(const float* __restrict__ in, _Float16* __restrict__ out, int n) {
  int i = blockIdx.x * 256 + threadIdx.x;
  if (i < n) out[i] = (_Float16)in[i];
}

// ---------------------------------------------------------------------------
// Generic WMMA GEMM:  C[M,O] = epi( A[M,K] @ W[O,K]^T + bias[O] )
// Block: 256 threads (8 wave32), tile 128(M) x 128(O), K-step 32.
// Waves in 4x2 grid; each wave computes 32x64 via 2x4 16x16 WMMA frags.
// Tiles staged via GLOBAL_LOAD_ASYNC_TO_LDS_B128.
// ---------------------------------------------------------------------------
enum { EPI_F16 = 0, EPI_RELU = 1, EPI_VT = 2, EPI_PROJ = 3, EPI_MIX = 4, EPI_F32 = 5 };

template <int EPI>
__global__ __launch_bounds__(256) void gemm_wmma(
    const _Float16* __restrict__ A, int lda,
    const _Float16* __restrict__ W, int K,
    const float* __restrict__ bias,
    void* __restrict__ Cout, int ldc,
    _Float16* __restrict__ aux0, _Float16* __restrict__ aux1) {
  __shared__ _Float16 As[128][48];  // +16 halves pad: rows stay 32B aligned
  __shared__ _Float16 Bs[128][48];

  const int tid = threadIdx.x, lane = tid & 31, wave = tid >> 5;
  const int wm = wave & 3, wo = wave >> 2;
  const int m0 = blockIdx.x * 128, o0 = blockIdx.y * 128;
  const int kh = (lane >> 4) * 8;     // unused (A and B frags both use sh16)
  const int sh16 = (lane >> 4) * 16;  // B-frag K base per lane-half
  const int lr = lane & 15;

  vf8 acc[2][4];
#pragma unroll
  for (int mi = 0; mi < 2; ++mi)
#pragma unroll
    for (int oi = 0; oi < 4; ++oi) acc[mi][oi] = vzero8();

  for (int k0 = 0; k0 < K; k0 += 32) {
    __syncthreads();
    // Stage A,B tiles (each 128x32 halves = 512 x 16B chunks): async DMA to LDS
#pragma unroll
    for (int c = tid; c < 512; c += 256) {
      int r = c >> 2, col = (c & 3) << 3;
      async_load_b128((uint32_t)(uintptr_t)&As[r][col],
                      A + (size_t)(m0 + r) * lda + k0 + col);
      async_load_b128((uint32_t)(uintptr_t)&Bs[r][col],
                      W + (size_t)(o0 + r) * K + k0 + col);
    }
    wait_asynccnt0();
    __syncthreads();

    vh16 a[2], b[4];
#pragma unroll
    for (int mi = 0; mi < 2; ++mi) {
      const _Float16* p = &As[wm * 32 + mi * 16 + lr][0];
      a[mi] = mkfrag(*(const vh8*)(p + sh16), *(const vh8*)(p + 8 + sh16));
    }
#pragma unroll
    for (int oi = 0; oi < 4; ++oi)
      b[oi] = *(const vh16*)(&Bs[wo * 64 + oi * 16 + lr][sh16]);
#pragma unroll
    for (int mi = 0; mi < 2; ++mi)
#pragma unroll
      for (int oi = 0; oi < 4; ++oi)
        acc[mi][oi] = WMMA_F16(a[mi], b[oi], acc[mi][oi]);
  }

  // Epilogue. C-frag layout: lane l -> n = l%16, rows m = (l/16)*8 + r
  const int rb = (lane >> 4) * 8;
#pragma unroll
  for (int mi = 0; mi < 2; ++mi) {
#pragma unroll
    for (int oi = 0; oi < 4; ++oi) {
      const int go = o0 + wo * 64 + oi * 16 + lr;
      const float bval = bias[go];
#pragma unroll
      for (int r = 0; r < 8; ++r) {
        const int gm = m0 + wm * 32 + mi * 16 + rb + r;
        float cv = acc[mi][oi][r] + bval;
        if constexpr (EPI == EPI_F16) {
          ((_Float16*)Cout)[(size_t)gm * ldc + go] = (_Float16)cv;
        } else if constexpr (EPI == EPI_RELU) {
          ((_Float16*)Cout)[(size_t)gm * ldc + go] = (_Float16)fmaxf(cv, 0.0f);
        } else if constexpr (EPI == EPI_VT) {
          // v stored transposed per batch: vt[batch][d=go][n=gm%2048]
          int batch = gm >> 11, nb = gm & 2047;
          ((_Float16*)Cout)[(size_t)batch * (128 * 2048) + (size_t)go * 2048 + nb] =
              (_Float16)cv;
        } else if constexpr (EPI == EPI_PROJ) {
          // projection f_b -> gate_in self-half + two `others` slots
          _Float16 hv = (_Float16)cv;
          size_t idx = (size_t)gm * 256 + go;
          ((_Float16*)Cout)[idx] = hv;
          aux0[idx] = hv;
          aux1[idx] = hv;
        } else if constexpr (EPI == EPI_MIX) {
          // gate MLP2 epilogue: sigmoid gate, mix f_self / attn, f32 out
          float sg = 1.0f / (1.0f + __expf(-cv));
          float fs = (float)aux0[(size_t)gm * 256 + go];
          float at = (float)aux0[(size_t)gm * 256 + 128 + go];
          ((float*)Cout)[(size_t)gm * ldc + go] = (1.0f - sg) * fs + sg * at;
        } else {  // EPI_F32
          ((float*)Cout)[(size_t)gm * ldc + go] = cv;
        }
      }
    }
  }
}

// ---------------------------------------------------------------------------
// Flash attention: per (branch, batch), d=128, N=2048 keys streamed in 32-chunks.
// Block = 128 threads = 4 wave32; each wave owns 16 query rows.
// Writes attn into gate_in[:, 128:256] (f16) for the gate MLP.
// ---------------------------------------------------------------------------
__global__ __launch_bounds__(128) void flash_attn(
    const _Float16* __restrict__ q,   // [3][16384][128]
    const _Float16* __restrict__ k,   // [3][16384][128]
    const _Float16* __restrict__ vt,  // [3][8][128][2048]
    _Float16* __restrict__ gate_in) { // [3][16384][256]
  __shared__ _Float16 P[4][16][48];  // per-wave P tile (16 q x 32 keys)

  const int lane = threadIdx.x & 31, wave = threadIdx.x >> 5;
  const int br = blockIdx.z, batch = blockIdx.y;
  const int m0 = blockIdx.x * 64 + wave * 16;  // query base within batch
  const size_t rowbase = (size_t)br * 16384 + (size_t)batch * 2048;
  const int kh = (lane >> 4) * 8, lr = lane & 15, sh16 = (lane >> 4) * 16;

  // q A-frags for this wave's 16 rows (K = d = 128 -> 4 frags of K32)
  vh16 qf[4];
  {
    const _Float16* qr = q + (rowbase + m0 + lr) * 128;
#pragma unroll
    for (int t = 0; t < 4; ++t)
      qf[t] = mkfrag(*(const vh8*)(qr + t * 32 + sh16),
                     *(const vh8*)(qr + t * 32 + 8 + sh16));
  }

  vf8 acc[8];
  float mmax[8], lsum[8];
#pragma unroll
  for (int i = 0; i < 8; ++i) {
    acc[i] = vzero8();
    mmax[i] = -1e30f;
    lsum[i] = 0.0f;
  }
  const float scale = 0.08838834764831845f;  // 1/sqrt(128)
  const _Float16* kbase = k + rowbase * 128;
  const _Float16* vbase = vt + ((size_t)br * 8 + batch) * (128 * 2048);

  for (int kc = 0; kc < 2048; kc += 32) {
    // scores: 16 queries x 32 keys, contracted over d=128
    vf8 s0 = vzero8(), s1 = vzero8();
#pragma unroll
    for (int t = 0; t < 4; ++t) {
      vh16 b0 = *(const vh16*)(kbase + (size_t)(kc + lr) * 128 + t * 32 + sh16);
      vh16 b1 = *(const vh16*)(kbase + (size_t)(kc + 16 + lr) * 128 + t * 32 + sh16);
      s0 = WMMA_F16(qf[t], b0, s0);
      s1 = WMMA_F16(qf[t], b1, s1);
    }

    // online softmax update (row stats replicated across 16 lanes per half)
    float alpha[8];
#pragma unroll
    for (int r = 0; r < 8; ++r) {
      float v0 = s0[r] * scale, v1 = s1[r] * scale;
      float cm = fmaxf(v0, v1);
      cm = fmaxf(cm, __shfl_xor(cm, 1));
      cm = fmaxf(cm, __shfl_xor(cm, 2));
      cm = fmaxf(cm, __shfl_xor(cm, 4));
      cm = fmaxf(cm, __shfl_xor(cm, 8));
      float nm = fmaxf(mmax[r], cm);
      alpha[r] = __expf(mmax[r] - nm);
      float p0 = __expf(v0 - nm), p1 = __expf(v1 - nm);
      float rs = p0 + p1;
      rs += __shfl_xor(rs, 1);
      rs += __shfl_xor(rs, 2);
      rs += __shfl_xor(rs, 4);
      rs += __shfl_xor(rs, 8);
      lsum[r] = lsum[r] * alpha[r] + rs;
      mmax[r] = nm;
      int pr = (lane >> 4) * 8 + r;  // row m of the P tile
      P[wave][pr][lr] = (_Float16)p0;
      P[wave][pr][16 + lr] = (_Float16)p1;
    }
#pragma unroll
    for (int dt = 0; dt < 8; ++dt)
#pragma unroll
      for (int r = 0; r < 8; ++r) acc[dt][r] *= alpha[r];

    // C-frag -> A-frag layout change for P via per-wave LDS region
    asm volatile("s_wait_dscnt 0" ::: "memory");
    vh16 pf = mkfrag(*(const vh8*)(&P[wave][lr][sh16]),
                     *(const vh8*)(&P[wave][lr][8 + sh16]));

    // acc += P(16x32) @ V(32x128); vt rows are contiguous in keys
#pragma unroll
    for (int dt = 0; dt < 8; ++dt) {
      vh16 bv = *(const vh16*)(vbase + (size_t)(dt * 16 + lr) * 2048 + kc + sh16);
      acc[dt] = WMMA_F16(pf, bv, acc[dt]);
    }
  }

  float inv[8];
#pragma unroll
  for (int r = 0; r < 8; ++r) inv[r] = 1.0f / lsum[r];
  _Float16* gout = gate_in + rowbase * 256;
#pragma unroll
  for (int dt = 0; dt < 8; ++dt)
#pragma unroll
    for (int r = 0; r < 8; ++r) {
      size_t gm = (size_t)(m0 + (lane >> 4) * 8 + r);
      gout[gm * 256 + 128 + dt * 16 + lr] = (_Float16)(acc[dt][r] * inv[r]);
    }
}

// ---------------------------------------------------------------------------
// LayerNorm over d=128; one wave per row; writes upd[16384][384] at branch col.
// ---------------------------------------------------------------------------
__global__ __launch_bounds__(256) void layernorm_kernel(
    const float* __restrict__ mixed,  // [3][16384][128]
    const float* __restrict__ g0, const float* __restrict__ b0,
    const float* __restrict__ g1, const float* __restrict__ b1,
    const float* __restrict__ g2, const float* __restrict__ b2,
    _Float16* __restrict__ upd) {  // [16384][384]
  const int lane = threadIdx.x & 31, wave = threadIdx.x >> 5;
  const int row = blockIdx.x * 8 + wave;  // 0..49151
  const int br = row >> 14, m = row & 16383;

  vf4 v = *(const vf4*)(mixed + (size_t)row * 128 + lane * 4);
  float s = v[0] + v[1] + v[2] + v[3];
  s += __shfl_xor(s, 1);
  s += __shfl_xor(s, 2);
  s += __shfl_xor(s, 4);
  s += __shfl_xor(s, 8);
  s += __shfl_xor(s, 16);
  float mu = s * (1.0f / 128.0f);
  float d0 = v[0] - mu, d1 = v[1] - mu, d2 = v[2] - mu, d3 = v[3] - mu;
  float sq = d0 * d0 + d1 * d1 + d2 * d2 + d3 * d3;
  sq += __shfl_xor(sq, 1);
  sq += __shfl_xor(sq, 2);
  sq += __shfl_xor(sq, 4);
  sq += __shfl_xor(sq, 8);
  sq += __shfl_xor(sq, 16);
  float rstd = rsqrtf(sq * (1.0f / 128.0f) + 1e-5f);

  const float* g = (br == 0) ? g0 : (br == 1) ? g1 : g2;
  const float* bb = (br == 0) ? b0 : (br == 1) ? b1 : b2;
  _Float16* o = upd + (size_t)m * 384 + br * 128 + lane * 4;
#pragma unroll
  for (int i = 0; i < 4; ++i) {
    int d = lane * 4 + i;
    o[i] = (_Float16)((v[i] - mu) * rstd * g[d] + bb[d]);
  }
}

// ---------------------------------------------------------------------------
// Host-side orchestration
// ---------------------------------------------------------------------------
extern "C" void kernel_launch(void* const* d_in, const int* in_sizes, int n_in,
                              void* d_out, int out_size, void* d_ws, size_t ws_size,
                              hipStream_t stream) {
  (void)in_sizes; (void)n_in; (void)out_size; (void)ws_size;
  const int BN = 16384;  // 8 * 2048

  const float* x = (const float*)d_in[0];
  const float *W_map[3], *B_map[3], *W_q[3], *B_q[3], *W_k[3], *B_k[3];
  const float *W_v[3], *B_v[3], *W_g1[3], *B_g1[3], *W_g2[3], *B_g2[3];
  const float *LN_g[3], *LN_b[3];
  for (int b = 0; b < 3; ++b) {
    int s = 1 + b * 14;
    W_map[b] = (const float*)d_in[s + 0];  B_map[b] = (const float*)d_in[s + 1];
    W_q[b]   = (const float*)d_in[s + 2];  B_q[b]   = (const float*)d_in[s + 3];
    W_k[b]   = (const float*)d_in[s + 4];  B_k[b]   = (const float*)d_in[s + 5];
    W_v[b]   = (const float*)d_in[s + 6];  B_v[b]   = (const float*)d_in[s + 7];
    W_g1[b]  = (const float*)d_in[s + 8];  B_g1[b]  = (const float*)d_in[s + 9];
    W_g2[b]  = (const float*)d_in[s + 10]; B_g2[b]  = (const float*)d_in[s + 11];
    LN_g[b]  = (const float*)d_in[s + 12]; LN_b[b]  = (const float*)d_in[s + 13];
  }
  const float* out_w = (const float*)d_in[43];
  const float* out_b = (const float*)d_in[44];

  // Workspace carve (deterministic bump allocator)
  uint8_t* wp = (uint8_t*)d_ws;
  auto alloc = [&](size_t bytes) -> uint8_t* {
    uint8_t* r = wp;
    wp += (bytes + 255) & ~(size_t)255;
    return r;
  };
  _Float16* xh = (_Float16*)alloc((size_t)BN * 512 * 2);
  _Float16 *mapw[3], *qw[3], *kw[3], *vw[3], *g1w[3], *g2w[3];
  for (int b = 0; b < 3; ++b) {
    mapw[b] = (_Float16*)alloc(128 * 512 * 2);
    qw[b]   = (_Float16*)alloc(128 * 128 * 2);
    kw[b]   = (_Float16*)alloc(128 * 256 * 2);
    vw[b]   = (_Float16*)alloc(128 * 256 * 2);
    g1w[b]  = (_Float16*)alloc(256 * 256 * 2);
    g2w[b]  = (_Float16*)alloc(128 * 256 * 2);
  }
  _Float16* outw   = (_Float16*)alloc(512 * 384 * 2);
  _Float16* gatein = (_Float16*)alloc((size_t)3 * BN * 256 * 2);  // [f_self | attn]
  _Float16* others = (_Float16*)alloc((size_t)3 * BN * 256 * 2);  // reused as h after k,v
  _Float16* qb     = (_Float16*)alloc((size_t)3 * BN * 128 * 2);
  _Float16* kb     = (_Float16*)alloc((size_t)3 * BN * 128 * 2);
  _Float16* vtb    = (_Float16*)alloc((size_t)3 * BN * 128 * 2);  // [3][8][128][2048]
  float*    mixed  = (float*)alloc((size_t)3 * BN * 128 * 4);
  _Float16* upd    = (_Float16*)alloc((size_t)BN * 384 * 2);

  auto cvt = [&](const float* src, _Float16* dst, int n) {
    f32_to_f16<<<(n + 255) / 256, 256, 0, stream>>>(src, dst, n);
  };
  cvt(x, xh, BN * 512);
  for (int b = 0; b < 3; ++b) {
    cvt(W_map[b], mapw[b], 128 * 512);
    cvt(W_q[b], qw[b], 128 * 128);
    cvt(W_k[b], kw[b], 128 * 256);
    cvt(W_v[b], vw[b], 128 * 256);
    cvt(W_g1[b], g1w[b], 256 * 256);
    cvt(W_g2[b], g2w[b], 128 * 256);
  }
  cvt(out_w, outw, 512 * 384);

  const size_t SZ = (size_t)BN * 256;
  _Float16* gib[3] = {gatein, gatein + SZ, gatein + 2 * SZ};
  _Float16* ob[3]  = {others, others + SZ, others + 2 * SZ};
  // f0 -> others1[:, :128], others2[:, :128]; f1 -> others0[:, :128], others2[:,128:];
  // f2 -> others0[:,128:], others1[:,128:]
  _Float16* pa[3][2] = {{ob[1], ob[2]}, {ob[0], ob[2] + 128}, {ob[0] + 128, ob[1] + 128}};

  // 1) input projections (K=512 -> 128), fan-out to gate_in self half + others
  for (int b = 0; b < 3; ++b)
    gemm_wmma<EPI_PROJ><<<dim3(128, 1), 256, 0, stream>>>(
        xh, 512, mapw[b], 512, B_map[b], gib[b], 256, pa[b][0], pa[b][1]);
  // 2) q (K=128), k (K=256), v (K=256, transposed store)
  for (int b = 0; b < 3; ++b)
    gemm_wmma<EPI_F16><<<dim3(128, 1), 256, 0, stream>>>(
        gib[b], 256, qw[b], 128, B_q[b], qb + (size_t)b * BN * 128, 128, nullptr, nullptr);
  for (int b = 0; b < 3; ++b)
    gemm_wmma<EPI_F16><<<dim3(128, 1), 256, 0, stream>>>(
        ob[b], 256, kw[b], 256, B_k[b], kb + (size_t)b * BN * 128, 128, nullptr, nullptr);
  for (int b = 0; b < 3; ++b)
    gemm_wmma<EPI_VT><<<dim3(128, 1), 256, 0, stream>>>(
        ob[b], 256, vw[b], 256, B_v[b], vtb + (size_t)b * BN * 128, 2048, nullptr, nullptr);
  // 3) flash attention -> gate_in[:, 128:256]
  flash_attn<<<dim3(32, 8, 3), 128, 0, stream>>>(qb, kb, vtb, gatein);
  // 4) gate MLP1 (relu) -> reuse `others` as h; MLP2 fused sigmoid-mix -> mixed f32
  for (int b = 0; b < 3; ++b)
    gemm_wmma<EPI_RELU><<<dim3(128, 2), 256, 0, stream>>>(
        gib[b], 256, g1w[b], 256, B_g1[b], ob[b], 256, nullptr, nullptr);
  for (int b = 0; b < 3; ++b)
    gemm_wmma<EPI_MIX><<<dim3(128, 1), 256, 0, stream>>>(
        ob[b], 256, g2w[b], 256, B_g2[b], mixed + (size_t)b * BN * 128, 128,
        gib[b], nullptr);
  // 5) layernorm -> upd f16 [16384][384]
  layernorm_kernel<<<6144, 256, 0, stream>>>(mixed, LN_g[0], LN_b[0], LN_g[1],
                                             LN_b[1], LN_g[2], LN_b[2], upd);
  // 6) output projection (K=384 -> 512) -> d_out f32
  gemm_wmma<EPI_F32><<<dim3(128, 4), 256, 0, stream>>>(
      upd, 384, outw, 384, out_b, d_out, 512, nullptr, nullptr);
}
